// ContinuousRNN_1752346657132
// MI455X (gfx1250) — compile-verified, matrix operand order checked by reading, not run on hardware
//
#include <hip/hip_runtime.h>
#include <hip/hip_bf16.h>

typedef __attribute__((ext_vector_type(16))) __bf16    v16bf;
typedef __attribute__((ext_vector_type(8)))  __bf16    v8bf;
typedef __attribute__((ext_vector_type(8)))  float     v8f;
typedef __attribute__((ext_vector_type(4)))  unsigned  v4u;
typedef __attribute__((ext_vector_type(8)))  unsigned  v8u;

#define BSZ   256
#define NSZ   2048
#define NSTEP 64
#define DTC   0.1f
#define KC    32
#define BM    64
#define BN_T  128
#define NITER (NSZ / KC)

// ---- CDNA5 Tensor Data Mover: 2D tile (rows x KC bf16, row stride NSZ elems)
// global -> LDS, one descriptor-driven DMA op, TENSORcnt-tracked.
// D# bitfields per CDNA5 ISA ch.8 (group0 128b, group1 256b; VADDR2/3 = NULL -> 2D).
__device__ __forceinline__ void tdm_load_2d(void* lds, const void* gaddr,
                                            unsigned rows_tile, unsigned rows_tensor) {
    unsigned           l = (unsigned)(size_t)(__attribute__((address_space(3))) void*)lds;
    unsigned long long g = (unsigned long long)(size_t)gaddr;
    v4u g0;
    g0[0] = 1u;                                            // count=1, user desc, no gather
    g0[1] = l;                                             // lds_addr (bytes)
    g0[2] = (unsigned)g;                                   // global_addr[31:0]
    g0[3] = (unsigned)((g >> 32) & 0x01FFFFFFu) | (2u << 30); // addr[56:32] | type=2
    v8u g1;
    g1[0] = (1u << 16);                                    // wg_mask=0, data_size=1 (2B)
    g1[1] = ((unsigned)NSZ & 0xFFFFu) << 16;               // tensor_dim0[15:0] @ 63:48
    g1[2] = ((unsigned)NSZ >> 16) | (rows_tensor << 16);   // dim0[31:16] | dim1[15:0]
    g1[3] = (rows_tensor >> 16) | ((unsigned)KC << 16);    // dim1[31:16] | tile_dim0
    g1[4] = rows_tile;                                     // tile_dim1 (tile_dim2=0)
    g1[5] = (unsigned)NSZ;                                 // tensor_dim0_stride[31:0]
    g1[6] = 0u;                                            // stride0 hi | stride1 lo
    g1[7] = 0u;                                            // stride1 hi
    asm volatile("tensor_load_to_lds %0, %1" :: "s"(g0), "s"(g1) : "memory");
}

// ---------------- once-per-launch: split fp32 weights into bf16 hi/lo planes
__global__ void split_weights(const float* __restrict__ W, const float* __restrict__ T,
                              __bf16* __restrict__ wHi, __bf16* __restrict__ wLo,
                              __bf16* __restrict__ tHi, __bf16* __restrict__ tLo) {
    long i = (long)blockIdx.x * blockDim.x + threadIdx.x;
    long NN = (long)NSZ * NSZ;
    if (i >= NN) return;
    float w = W[i];
    __bf16 wh = (__bf16)w;
    wHi[i] = wh;
    wLo[i] = (__bf16)(w - (float)wh);
    float t = T[i];
    __bf16 th = (__bf16)t;
    tHi[i] = th;
    tLo[i] = (__bf16)(t - (float)th);
}

// ---------------- step-0 only: build A-planes a = relu(s)*relu(x), psi = relu(p)
__global__ void act_split(const float* __restrict__ x, const float* __restrict__ s,
                          const float* __restrict__ p,
                          __bf16* __restrict__ aHi, __bf16* __restrict__ aLo,
                          __bf16* __restrict__ psHi, __bf16* __restrict__ psLo) {
    long i = (long)blockIdx.x * blockDim.x + threadIdx.x;
    float xv = x[i], sv = s[i], pv = p[i];
    float a   = fmaxf(sv, 0.f) * fmaxf(xv, 0.f);
    float psi = fmaxf(pv, 0.f);
    __bf16 ah = (__bf16)a;
    aHi[i] = ah;
    aLo[i] = (__bf16)(a - (float)ah);
    __bf16 ph = (__bf16)psi;
    psHi[i] = ph;
    psLo[i] = (__bf16)(psi - (float)ph);
}

// 16x32 bf16 WMMA A/B fragment from an LDS tile row (wave32 layout:
// lanes 0-15 hold K 0..7 & 16..23, lanes 16-31 hold K 8..15 & 24..31)
__device__ __forceinline__ v16bf ldfrag(const __bf16* row, int half) {
    v8bf lo = *(const v8bf*)(row + half * 8);
    v8bf hi = *(const v8bf*)(row + 16 + half * 8);
    return __builtin_shufflevector(lo, hi, 0,1,2,3,4,5,6,7,8,9,10,11,12,13,14,15);
}

// wave0: issue TDM DMA for one K-chunk (4 A-planes + 4 B-planes) into one buffer
__device__ __forceinline__ void stage_tdm(int mt, int nt, int k0,
                                          const __bf16* const* Apl,
                                          const __bf16* const* Bpl,
                                          __bf16 (*sAb)[BM][KC],
                                          __bf16 (*sBb)[BN_T][KC]) {
#pragma unroll
    for (int pl = 0; pl < 4; ++pl)
        tdm_load_2d(&sAb[pl][0][0], Apl[pl] + (long)(mt * BM) * NSZ + k0, BM, BSZ);
#pragma unroll
    for (int pl = 0; pl < 4; ++pl)
        tdm_load_2d(&sBb[pl][0][0], Bpl[pl] + (long)(nt * BN_T) * NSZ + k0, BN_T, NSZ);
}

// ---------------- fused dual-GEMM (bf16x3) + RNN update + next-step A-plane emit
__global__ __launch_bounds__(256, 2)
void rnn_step(const float* __restrict__ x_in, const float* __restrict__ s_in,
              const float* __restrict__ p_in,
              const __bf16* __restrict__ aHi, const __bf16* __restrict__ aLo,
              const __bf16* __restrict__ psHi, const __bf16* __restrict__ psLo,
              __bf16* __restrict__ aHiO, __bf16* __restrict__ aLoO,
              __bf16* __restrict__ psHiO, __bf16* __restrict__ psLoO,
              const __bf16* __restrict__ wHi, const __bf16* __restrict__ wLo,
              const __bf16* __restrict__ tHi, const __bf16* __restrict__ tLo,
              const float* __restrict__ Wb, const float* __restrict__ Tb,
              const float* __restrict__ wPtS, const float* __restrict__ wStP,
              const float* __restrict__ fri,
              float* __restrict__ x_out, float* __restrict__ s_out,
              float* __restrict__ p_out) {
    __shared__ __attribute__((aligned(16))) __bf16 sA[2][4][BM][KC];
    __shared__ __attribute__((aligned(16))) __bf16 sB[2][4][BN_T][KC];

    const int tid   = threadIdx.x;
    const int nt    = blockIdx.x;          // 0..15  column tile (128 cols)
    const int mt    = blockIdx.y;          // 0..3   row tile (64 rows)
    const int wave  = tid >> 5;
    const int half  = (tid >> 4) & 1;
    const int m16   = tid & 15;
    const int waveM = wave >> 2;           // 0..1
    const int waveN = wave & 3;            // 0..3
    const bool ldr  = (wave == 0);         // TDM issuing wave

    const __bf16* Apl[4] = { aHi, aLo, psHi, psLo };
    const __bf16* Bpl[4] = { wHi, wLo, tHi, tLo };

    v8f accX[2][2] = {};
    v8f accP[2][2] = {};

    // prologue: DMA buffer 0
    if (ldr) stage_tdm(mt, nt, 0, Apl, Bpl, sA[0], sB[0]);

    for (int ki = 0; ki < NITER; ++ki) {
        const int cur = ki & 1;
        if (ldr) __builtin_amdgcn_s_wait_tensorcnt(0); // my tiles for [cur] landed
        __syncthreads();                               // release implies tiles visible
        if (ki + 1 < NITER && ldr)                     // overlap next DMA with math
            stage_tdm(mt, nt, (ki + 1) * KC, Apl, Bpl, sA[cur ^ 1], sB[cur ^ 1]);

        v16bf fa[4][2], fb[4][2];
#pragma unroll
        for (int pl = 0; pl < 4; ++pl)
#pragma unroll
            for (int mi = 0; mi < 2; ++mi)
                fa[pl][mi] = ldfrag(&sA[cur][pl][waveM * 32 + mi * 16 + m16][0], half);
#pragma unroll
        for (int pl = 0; pl < 4; ++pl)
#pragma unroll
            for (int ni = 0; ni < 2; ++ni)
                fb[pl][ni] = ldfrag(&sB[cur][pl][waveN * 32 + ni * 16 + m16][0], half);

#pragma unroll
        for (int mi = 0; mi < 2; ++mi)
#pragma unroll
            for (int ni = 0; ni < 2; ++ni) {
                // bf16x3: hi*hi + hi*lo + lo*hi, fp32 accumulate
                accX[mi][ni] = __builtin_amdgcn_wmma_f32_16x16x32_bf16(
                    false, fa[0][mi], false, fb[0][ni], (short)0, accX[mi][ni], false, false);
                accX[mi][ni] = __builtin_amdgcn_wmma_f32_16x16x32_bf16(
                    false, fa[0][mi], false, fb[1][ni], (short)0, accX[mi][ni], false, false);
                accX[mi][ni] = __builtin_amdgcn_wmma_f32_16x16x32_bf16(
                    false, fa[1][mi], false, fb[0][ni], (short)0, accX[mi][ni], false, false);
                accP[mi][ni] = __builtin_amdgcn_wmma_f32_16x16x32_bf16(
                    false, fa[2][mi], false, fb[2][ni], (short)0, accP[mi][ni], false, false);
                accP[mi][ni] = __builtin_amdgcn_wmma_f32_16x16x32_bf16(
                    false, fa[2][mi], false, fb[3][ni], (short)0, accP[mi][ni], false, false);
                accP[mi][ni] = __builtin_amdgcn_wmma_f32_16x16x32_bf16(
                    false, fa[3][mi], false, fb[2][ni], (short)0, accP[mi][ni], false, false);
            }
    }

    // epilogue: C layout VGPR r, lanes 0-15 -> M=r, lanes 16-31 -> M=8+r
#pragma unroll
    for (int mi = 0; mi < 2; ++mi)
#pragma unroll
        for (int ni = 0; ni < 2; ++ni) {
            int col = nt * BN_T + waveN * 32 + ni * 16 + m16;
            float wb = Wb[col], tb = Tb[col];
            float a_pts = wPtS[col], a_stp = wStP[col];
#pragma unroll
            for (int r = 0; r < 8; ++r) {
                int row = mt * BM + waveM * 32 + mi * 16 + half * 8 + r;
                long idx = (long)row * NSZ + col;
                float xv = x_in[idx], sv = s_in[idx], pv = p_in[idx], fv = fri[idx];
                float phi = fmaxf(xv, 0.f), g = fmaxf(sv, 0.f), psi = fmaxf(pv, 0.f);
                float dx = -xv + accX[mi][ni][r] + wb;
                float ds = -sv + a_pts * psi + phi;
                float dp = -pv + accP[mi][ni][r] + tb + a_stp * g;
                float xn = xv + DTC * dx * fv;
                float sn = pv + DTC * ds;   // note: base is p, as in reference
                float pn = pv + DTC * dp;
                x_out[idx] = xn;
                s_out[idx] = sn;
                p_out[idx] = pn;
                // emit next step's bf16 hi/lo A-planes (ping-ponged buffers)
                float an  = fmaxf(sn, 0.f) * fmaxf(xn, 0.f);
                float pln = fmaxf(pn, 0.f);
                __bf16 ah = (__bf16)an;
                aHiO[idx] = ah;
                aLoO[idx] = (__bf16)(an - (float)ah);
                __bf16 ph = (__bf16)pln;
                psHiO[idx] = ph;
                psLoO[idx] = (__bf16)(pln - (float)ph);
            }
        }
}

extern "C" void kernel_launch(void* const* d_in, const int* in_sizes, int n_in,
                              void* d_out, int out_size, void* d_ws, size_t ws_size,
                              hipStream_t stream) {
    (void)in_sizes; (void)n_in; (void)out_size; (void)ws_size;
    const float* x0   = (const float*)d_in[0];
    const float* fri  = (const float*)d_in[1];
    const float* Ww   = (const float*)d_in[2];
    const float* Wb   = (const float*)d_in[3];
    const float* Tw   = (const float*)d_in[4];
    const float* Tb   = (const float*)d_in[5];
    const float* wPtS = (const float*)d_in[6];
    const float* wStP = (const float*)d_in[7];
    float* out = (float*)d_out;              // [x (B*N) | xs (STEPS*B*N)]
    char*  ws  = (char*)d_ws;

    const long NN = (long)NSZ * NSZ;
    const long BN = (long)BSZ * NSZ;

    __bf16* wHi = (__bf16*)(ws);
    __bf16* wLo = (__bf16*)(ws + 2 * NN);
    __bf16* tHi = (__bf16*)(ws + 4 * NN);
    __bf16* tLo = (__bf16*)(ws + 6 * NN);
    // 2 ping-pong sets of 4 activation planes (bf16, B*N each)
    __bf16* pln[2][4];
    for (int s2 = 0; s2 < 2; ++s2)
        for (int k = 0; k < 4; ++k)
            pln[s2][k] = (__bf16*)(ws + 8 * NN + (long)(s2 * 4 + k) * 2 * BN);
    float* sB0 = (float*)(ws + 8 * NN + 16 * BN);
    float* sB1 = (float*)(ws + 8 * NN + 20 * BN);
    float* pB0 = (float*)(ws + 8 * NN + 24 * BN);
    float* pB1 = (float*)(ws + 8 * NN + 28 * BN);

    split_weights<<<(int)((NN + 255) / 256), 256, 0, stream>>>(Ww, Tw, wHi, wLo, tHi, tLo);
    act_split<<<(int)(BN / 256), 256, 0, stream>>>(x0, x0, x0,
                                                   pln[0][0], pln[0][1], pln[0][2], pln[0][3]);

    for (int t = 0; t < NSTEP; ++t) {
        const float* x_in = (t == 0) ? x0 : out + BN + (long)(t - 1) * BN;
        const float* s_in = (t == 0) ? x0 : (((t - 1) & 1) ? sB1 : sB0);
        const float* p_in = (t == 0) ? x0 : (((t - 1) & 1) ? pB1 : pB0);
        float* s_out = (t & 1) ? sB1 : sB0;
        float* p_out = (t & 1) ? pB1 : pB0;
        float* x_out = out + BN + (long)t * BN;
        const int ci = t & 1, no = (t + 1) & 1;

        rnn_step<<<dim3(NSZ / BN_T, BSZ / BM), 256, 0, stream>>>(
            x_in, s_in, p_in,
            pln[ci][0], pln[ci][1], pln[ci][2], pln[ci][3],
            pln[no][0], pln[no][1], pln[no][2], pln[no][3],
            wHi, wLo, tHi, tLo, Wb, Tb, wPtS, wStP, fri,
            x_out, s_out, p_out);
    }
    // final x = xs[STEPS-1]
    hipMemcpyAsync(out, out + BN + (long)(NSTEP - 1) * BN, BN * sizeof(float),
                   hipMemcpyDeviceToDevice, stream);
}
